// MultiHeadForgettingAttention_74775380623507
// MI455X (gfx1250) — matrix operand order repaired and checked by reading, not compiled, not run on hardware
//
#include <hip/hip_runtime.h>
#include <hip/hip_bf16.h>

typedef __attribute__((ext_vector_type(16))) _Float16     v16h;
typedef __attribute__((ext_vector_type(8)))  float        v8f;
typedef __attribute__((ext_vector_type(8)))  _Float16     h8;
typedef __attribute__((ext_vector_type(4)))  _Float16     h4;
typedef __attribute__((ext_vector_type(4)))  unsigned int v4u;
typedef __attribute__((ext_vector_type(8)))  int          v8i;
typedef __attribute__((ext_vector_type(4)))  int          v4i;

#define DEV static __device__ __forceinline__

constexpr int BB  = 2;
constexpr int SS  = 2048;
constexpr int DD  = 1024;
constexpr int HH  = 16;
constexpr int DKK = 64;
constexpr int MTOT = BB * SS;   // 4096 rows in the (B*S, D) GEMM view

#if __has_builtin(__builtin_amdgcn_tensor_load_to_lds) && \
    __has_builtin(__builtin_amdgcn_s_wait_tensorcnt)
#define USE_TDM 1
#else
#define USE_TDM 0
#endif

// ---------------------------------------------------------------------------
// WMMA helpers (wave32, v_wmma_f32_16x16x32_f16)
// A (16x32 f16): lane -> m = lane&15, half = lane>>4; VGPR v packs K pair at
//   k = 2*(v&3) + 16*(v>>2) + 8*half          (ISA 7.12.2 16-bit A layout)
// B (32x16 f16): lane -> n = lane&15; VGPR v packs K pair (k, k+1), k as in A
// C/D (16x16 f32): lane -> n = lane&15; VGPR r -> m = r + 8*(lane>>4)
// ---------------------------------------------------------------------------
DEV v8f wmma(v16h a, v16h b, v8f c) {
    return __builtin_amdgcn_wmma_f32_16x16x32_f16(false, a, false, b,
                                                  (short)0, c, false, false);
}

// A fragment from a row-major (m, k) tile, row stride ld elements
DEV v16h frag_a(const _Float16* t, int ld) {
    int lane = threadIdx.x & 31;
    int m = lane & 15, half = lane >> 4;
    v16h a;
#pragma unroll
    for (int v = 0; v < 8; ++v) {
        int k = 2 * (v & 3) + 16 * (v >> 2) + 8 * half;
        a[2 * v]     = t[m * ld + k];
        a[2 * v + 1] = t[m * ld + k + 1];
    }
    return a;
}

// B fragment from (k, n) row-major storage: lane -> n = lane&15, K pairs as in frag_a
DEV v16h frag_b_kn(const _Float16* t, int ld) {
    int lane = threadIdx.x & 31;
    int n = lane & 15, half = lane >> 4;
    v16h b;
#pragma unroll
    for (int v = 0; v < 8; ++v) {
        int k = 2 * (v & 3) + 16 * (v >> 2) + 8 * half;
        b[2 * v]     = t[k * ld + n];
        b[2 * v + 1] = t[(k + 1) * ld + n];
    }
    return b;
}

// ---------------------------------------------------------------------------
// Tensor Data Mover: DMA `rows` rows of 64 f16 (row pitch 64 elem in memory)
// into LDS with row pitch 72 elem (pad_interval=32 DW, pad_amount=4 DW).
// ---------------------------------------------------------------------------
DEV void tdm_load_rows_64h(const _Float16* gsrc, unsigned lds_byte_off,
                           int rows) {
#if USE_TDM
    unsigned long long ga = (unsigned long long)(const void*)gsrc;
    v4u g0;
    g0[0] = 1u;                                      // count=1, user D#
    g0[1] = lds_byte_off;                            // lds_addr
    g0[2] = (unsigned)(ga & 0xffffffffu);            // global_addr[31:0]
    g0[3] = (unsigned)((ga >> 32) & 0x01ffffffu)     // global_addr[56:32]
          | (2u << 30);                              // type = 2 (image)
    v8i g1;
    g1[0] = (1 << 16)                                // data_size = 2 bytes
          | (1 << 20)                                // pad_enable
          | (4 << 22)                                // pad_interval: 32 DW
          | (3 << 25);                               // pad_amount: 4 DW
    g1[1] = (DKK & 0xffff) << 16;                    // tensor_dim0 = 64
    g1[2] = (SS & 0xffff) << 16;                     // tensor_dim1 = 2048
    g1[3] = (DKK & 0xffff) << 16;                    // tile_dim0 = 64
    g1[4] = rows & 0xffff;                           // tile_dim1, tile_dim2=0
    g1[5] = DKK;                                     // dim0_stride lo32 = 64
    g1[6] = (DKK & 0xffff) << 16;                    // dim1_stride lo16 = 64
    g1[7] = 0;
    v4i gz = {0, 0, 0, 0};
#if defined(__clang_major__) && __clang_major__ >= 23
    v8i gz8 = {0, 0, 0, 0, 0, 0, 0, 0};
    __builtin_amdgcn_tensor_load_to_lds(g0, g1, gz, gz, gz8, 0);
#else
    __builtin_amdgcn_tensor_load_to_lds(g0, g1, gz, gz, 0);
#endif
#else
    (void)gsrc; (void)lds_byte_off; (void)rows;
#endif
}

// ---------------------------------------------------------------------------
// GEMM: Y[m,n] = sum_k X[m,k] * W[n,k] + bias[n]    (M=4096, N=K=1024)
// A_F16:   X is _Float16 (else float, converted on load)
// OUT_PERM: store f16 into (B,H,S,DK) layout (else f32 into (M,N))
// 256 threads / 8 waves, 128x128 tile, K-step 32.
// W tile staged TRANSPOSED (k,n) so B fragments are contiguous LDS reads.
// ---------------------------------------------------------------------------
template <bool A_F16, bool OUT_PERM>
__global__ __launch_bounds__(256) void gemm_xwT(const void* __restrict__ Xv,
                                                const float* __restrict__ W,
                                                const float* __restrict__ bias,
                                                void* __restrict__ outv) {
    __shared__ _Float16 a_lds[128 * 40];   // (m, k) pitch 40
    __shared__ _Float16 bT[32 * 136];      // (k, n) pitch 136

    const int m_blk = blockIdx.y * 128;
    const int n_blk = blockIdx.x * 128;
    const int tid   = threadIdx.x;
    const int wave  = tid >> 5;
    const int lane  = tid & 31;
    const int nl    = lane & 15;
    const int half  = lane >> 4;
    const int m_off = (wave >> 1) * 32;
    const int n_off = (wave & 1) * 64;

    const float*    Xf = (const float*)Xv;
    const _Float16* Xh = (const _Float16*)Xv;

    v8f acc[2][4] = {};

    for (int k0 = 0; k0 < DD; k0 += 32) {
        // ---- stage A (128x32) vectorized
        if (A_F16) {
#pragma unroll
            for (int t = 0; t < 2; ++t) {
                int i = tid * 8 + t * 2048;
                int r = i >> 5, c = i & 31;
                h8 x = *(const h8*)&Xh[(size_t)(m_blk + r) * DD + k0 + c];
                h4 lo = {x[0], x[1], x[2], x[3]};
                h4 hi = {x[4], x[5], x[6], x[7]};
                *(h4*)&a_lds[r * 40 + c]     = lo;
                *(h4*)&a_lds[r * 40 + c + 4] = hi;
            }
        } else {
#pragma unroll
            for (int t = 0; t < 4; ++t) {
                int i = tid * 4 + t * 1024;
                int r = i >> 5, c = i & 31;
                float4 x = *(const float4*)&Xf[(size_t)(m_blk + r) * DD + k0 + c];
                h4 hv = {(_Float16)x.x, (_Float16)x.y,
                         (_Float16)x.z, (_Float16)x.w};
                *(h4*)&a_lds[r * 40 + c] = hv;
            }
        }
        // ---- stage W tile transposed into (k, n)
#pragma unroll
        for (int t = 0; t < 4; ++t) {
            int i = tid * 4 + t * 1024;
            int r = i >> 5, c = i & 31;   // r = n local, c = k local
            float4 w4 = *(const float4*)&W[(size_t)(n_blk + r) * DD + k0 + c];
            bT[(c + 0) * 136 + r] = (_Float16)w4.x;
            bT[(c + 1) * 136 + r] = (_Float16)w4.y;
            bT[(c + 2) * 136 + r] = (_Float16)w4.z;
            bT[(c + 3) * 136 + r] = (_Float16)w4.w;
        }
        if (k0 + 32 < DD) {
            __builtin_prefetch((const char*)Xv +
                               ((size_t)(m_blk + (tid >> 3)) * DD + k0 + 32) *
                                   (A_F16 ? 2 : 4), 0, 3);
        }
        __syncthreads();

        v16h a0 = frag_a(&a_lds[(m_off)      * 40], 40);
        v16h a1 = frag_a(&a_lds[(m_off + 16) * 40], 40);
#pragma unroll
        for (int ni = 0; ni < 4; ++ni) {
            v16h b = frag_b_kn(&bT[n_off + 16 * ni], 136);
            acc[0][ni] = wmma(a0, b, acc[0][ni]);
            acc[1][ni] = wmma(a1, b, acc[1][ni]);
        }
        __syncthreads();
    }

    // ---- epilogue: bias add + store
#pragma unroll
    for (int mi = 0; mi < 2; ++mi) {
#pragma unroll
        for (int ni = 0; ni < 4; ++ni) {
            int n = n_blk + n_off + 16 * ni + nl;
            float bvn = bias[n];
#pragma unroll
            for (int r = 0; r < 8; ++r) {
                int m = m_blk + m_off + 16 * mi + r + 8 * half;
                float y = acc[mi][ni][r] + bvn;
                if (OUT_PERM) {
                    int b = m >> 11, s = m & (SS - 1);
                    int h = n >> 6,  dk = n & (DKK - 1);
                    ((_Float16*)outv)[(((size_t)b * HH + h) * SS + s) * DKK + dk] =
                        (_Float16)y;
                } else {
                    ((float*)outv)[(size_t)m * DD + n] = y;
                }
            }
        }
    }
}

// ---------------------------------------------------------------------------
// Forget-gate row dots: gq[i] = Q[i,:].wgq , gk[i] = K[i,:].wgk
// ---------------------------------------------------------------------------
__global__ __launch_bounds__(256) void gate_kernel(
    const _Float16* __restrict__ Q, const _Float16* __restrict__ K,
    const float* __restrict__ wgq, const float* __restrict__ wgk,
    float* __restrict__ gq, float* __restrict__ gk) {
    int i = blockIdx.x * blockDim.x + threadIdx.x;
    if (i >= BB * HH * SS) return;
    const _Float16* qr = Q + (size_t)i * DKK;
    const _Float16* kr = K + (size_t)i * DKK;
    float a = 0.f, b = 0.f;
#pragma unroll
    for (int d0 = 0; d0 < DKK; d0 += 8) {
        h8 qv = *(const h8*)&qr[d0];
        h8 kv = *(const h8*)&kr[d0];
#pragma unroll
        for (int u = 0; u < 8; ++u) {
            a += (float)qv[u] * wgq[d0 + u];
            b += (float)kv[u] * wgk[d0 + u];
        }
    }
    gq[i] = a;
    gk[i] = b;
}

// ---------------------------------------------------------------------------
// Gated causal flash attention.
// grid = (S/128, B*H); 256 threads / 8 waves; each wave owns 16 query rows.
// Uniform key-chunk loop over the whole block so all waves cooperatively
// stage K (transposed, dk-major) and V (TDM async DMA) chunks into LDS.
// ---------------------------------------------------------------------------
__global__ __launch_bounds__(256) void attn_kernel(
    const _Float16* __restrict__ Q, const _Float16* __restrict__ K,
    const _Float16* __restrict__ V, const float* __restrict__ gq,
    const float* __restrict__ gk, const float* __restrict__ gbp,
    _Float16* __restrict__ O) {
    __shared__ _Float16 kT[64 * 40];         // (dk, j) pitch 40
    __shared__ _Float16 vt[32 * 72];         // (j, dk) pitch 72 (TDM-padded)
    __shared__ _Float16 p_lds[8][16 * 34];   // per-wave P tile

    const int bh   = blockIdx.y;
    const int b    = bh / HH;
    const int h    = bh % HH;
    const int tid  = threadIdx.x;
    const int wave = tid >> 5;
    const int lane = tid & 31;
    const int nl   = lane & 15;
    const int half = lane >> 4;
    const int q_blk = blockIdx.x * 128;
    const int q0    = q_blk + wave * 16;

    const _Float16* Qh = Q + (size_t)bh * SS * DKK;
    const _Float16* Kh = K + (size_t)bh * SS * DKK;
    const _Float16* Vh = V + (size_t)bh * SS * DKK;
    const float*    gqh = gq + (size_t)bh * SS;
    const float*    gkh = gk + (size_t)bh * SS;
    const float     gbias = gbp[0];
    const float     scale = 0.125f;   // 1/sqrt(64)

    v16h qa0 = frag_a(Qh + (size_t)q0 * DKK,      DKK);
    v16h qa1 = frag_a(Qh + (size_t)q0 * DKK + 32, DKK);

    float gq_r[8];
#pragma unroll
    for (int r = 0; r < 8; ++r) gq_r[r] = gqh[q0 + r + 8 * half];

    float m_r[8], l_r[8];
#pragma unroll
    for (int r = 0; r < 8; ++r) { m_r[r] = -INFINITY; l_r[r] = 0.f; }
    v8f acc[4] = {};

    _Float16* pt = &p_lds[wave][0];
#if USE_TDM
    const unsigned vt_off = (unsigned)(size_t)(void*)&vt[0];
#endif

    const int jmax = q_blk + 127;   // last query row owned by this block
    for (int j0 = 0; j0 <= jmax; j0 += 32) {
        // ---- stage K chunk transposed to (dk, j): one h8 per thread
        {
            int i = tid * 8;                  // 32*64 = 2048 halfs
            int j = i >> 6, c = i & 63;
            h8 kv = *(const h8*)&Kh[(size_t)(j0 + j) * DKK + c];
#pragma unroll
            for (int u = 0; u < 8; ++u) kT[(c + u) * 40 + j] = kv[u];
        }
        // ---- stage V chunk (j, dk): TDM async DMA, 4 rows per wave
#if USE_TDM
        tdm_load_rows_64h(Vh + (size_t)(j0 + wave * 4) * DKK,
                          vt_off + (unsigned)(wave * 4 * 72 * 2), 4);
        __builtin_amdgcn_s_wait_tensorcnt(0);
#else
        {
            int i = tid * 8;
            int j = i >> 6, c = i & 63;
            h8 vv = *(const h8*)&Vh[(size_t)(j0 + j) * DKK + c];
            h4 lo = {vv[0], vv[1], vv[2], vv[3]};
            h4 hi = {vv[4], vv[5], vv[6], vv[7]};
            *(h4*)&vt[j * 72 + c]     = lo;
            *(h4*)&vt[j * 72 + c + 4] = hi;
        }
#endif
        __syncthreads();

        // ---- scores for two 16-wide key subtiles
        v8f sc[2];
#pragma unroll
        for (int sub = 0; sub < 2; ++sub) {
            v8f c = {};
            v16h bk0 = frag_b_kn(&kT[16 * sub],           40);  // dk 0..31
            v16h bk1 = frag_b_kn(&kT[32 * 40 + 16 * sub], 40);  // dk 32..63
            c = wmma(qa0, bk0, c);
            c = wmma(qa1, bk1, c);
            sc[sub] = c;
        }

        float gk0 = gkh[j0 + nl];
        float gk1 = gkh[j0 + 16 + nl];

        // ---- gate, causal mask, row-max
        float rowmax[8];
#pragma unroll
        for (int r = 0; r < 8; ++r) {
            int qi = q0 + r + 8 * half;
            float best = -INFINITY;
#pragma unroll
            for (int sub = 0; sub < 2; ++sub) {
                int j = j0 + 16 * sub + nl;
                float gl = gq_r[r] + (sub ? gk1 : gk0) + gbias;
                float gate = 1.f / (1.f + __expf(-gl));
                float val = (j <= qi) ? sc[sub][r] * scale * gate : -INFINITY;
                sc[sub][r] = val;
                best = fmaxf(best, val);
            }
#pragma unroll
            for (int off = 1; off < 16; off <<= 1)
                best = fmaxf(best, __shfl_xor(best, off, 32));
            rowmax[r] = best;
        }

        // ---- online softmax update + P tile
#pragma unroll
        for (int r = 0; r < 8; ++r) {
            float mn   = fmaxf(m_r[r], rowmax[r]);
            float corr = __expf(m_r[r] - mn);   // first chunk: exp(-inf)=0
            m_r[r] = mn;
            float rsum = 0.f;
#pragma unroll
            for (int sub = 0; sub < 2; ++sub) {
                float p = __expf(sc[sub][r] - mn);   // masked -> 0
                sc[sub][r] = p;
                rsum += p;
            }
#pragma unroll
            for (int off = 1; off < 16; off <<= 1)
                rsum += __shfl_xor(rsum, off, 32);
            l_r[r] = l_r[r] * corr + rsum;
#pragma unroll
            for (int ni = 0; ni < 4; ++ni) acc[ni][r] *= corr;
            pt[(r + 8 * half) * 34 + nl]      = (_Float16)sc[0][r];
            pt[(r + 8 * half) * 34 + 16 + nl] = (_Float16)sc[1][r];
        }

        // ---- P (16x32) @ V (32x64)
        v16h pa = frag_a(pt, 34);
#pragma unroll
        for (int ni = 0; ni < 4; ++ni) {
            v16h bv = frag_b_kn(&vt[16 * ni], 72);
            acc[ni] = wmma(pa, bv, acc[ni]);
        }
        __syncthreads();   // before next chunk overwrites kT / vt
    }

    // ---- normalize + store to (B, S, H*DK) f16
#pragma unroll
    for (int r = 0; r < 8; ++r) {
        int qi = q0 + r + 8 * half;
        float inv = 1.f / l_r[r];
        _Float16* orow = O + ((size_t)b * SS + qi) * DD + h * DKK;
#pragma unroll
        for (int ni = 0; ni < 4; ++ni)
            orow[16 * ni + nl] = (_Float16)(acc[ni][r] * inv);
    }
}

// ---------------------------------------------------------------------------
extern "C" void kernel_launch(void* const* d_in, const int* in_sizes, int n_in,
                              void* d_out, int out_size, void* d_ws,
                              size_t ws_size, hipStream_t stream) {
    (void)in_sizes; (void)n_in; (void)out_size; (void)ws_size;

    const float* q_in = (const float*)d_in[0];
    const float* k_in = (const float*)d_in[1];
    const float* v_in = (const float*)d_in[2];
    // d_in[3]: causal mask — causality computed analytically in-kernel
    const float* Wq = (const float*)d_in[4];
    const float* Wk = (const float*)d_in[5];
    const float* Wv = (const float*)d_in[6];
    const float* Wo = (const float*)d_in[7];
    const float* bq = (const float*)d_in[8];
    const float* bk = (const float*)d_in[9];
    const float* bv = (const float*)d_in[10];
    const float* bo = (const float*)d_in[11];
    const float* wgq = (const float*)d_in[12];
    const float* wgk = (const float*)d_in[13];
    const float* gb  = (const float*)d_in[14];

    char* ws = (char*)d_ws;
    size_t off = 0;
    auto wsalloc = [&](size_t bytes) -> void* {
        void* p = ws + off;
        off += (bytes + 255) & ~(size_t)255;
        return p;
    };
    const size_t nElem = (size_t)MTOT * DD;
    _Float16* Qh  = (_Float16*)wsalloc(nElem * 2);   // (B,H,S,DK)
    _Float16* Kh  = (_Float16*)wsalloc(nElem * 2);
    _Float16* Vh  = (_Float16*)wsalloc(nElem * 2);
    _Float16* AOh = (_Float16*)wsalloc(nElem * 2);   // (B,S,D)
    float*    gqb = (float*)wsalloc((size_t)BB * HH * SS * 4);
    float*    gkb = (float*)wsalloc((size_t)BB * HH * SS * 4);

    dim3 gg(DD / 128, MTOT / 128);   // (8, 32)

    gemm_xwT<false, true><<<gg, 256, 0, stream>>>(q_in, Wq, bq, Qh);
    gemm_xwT<false, true><<<gg, 256, 0, stream>>>(k_in, Wk, bk, Kh);
    gemm_xwT<false, true><<<gg, 256, 0, stream>>>(v_in, Wv, bv, Vh);

    gate_kernel<<<(BB * HH * SS) / 256, 256, 0, stream>>>(Qh, Kh, wgq, wgk,
                                                          gqb, gkb);

    attn_kernel<<<dim3(SS / 128, BB * HH), 256, 0, stream>>>(Qh, Kh, Vh, gqb,
                                                             gkb, gb, AOh);

    gemm_xwT<true, false><<<gg, 256, 0, stream>>>(AOh, Wo, bo, (float*)d_out);
}
